// Block_4724464025953
// MI455X (gfx1250) — compile-verified, matrix operand order checked by reading, not run on hardware
//
#include <hip/hip_runtime.h>

typedef _Float16 half_t;
typedef __attribute__((ext_vector_type(16))) _Float16 v16h;
typedef __attribute__((ext_vector_type(8)))  _Float16 v8h;
typedef __attribute__((ext_vector_type(8)))  float    v8f;

#define QN 25600   // H*W queries
#define CN 256     // embed dims
#define HGRID 160
#define WGRID 160
#define SMB_STRIDE 264   // 256 + 8 halves pad -> 528B row pitch, spreads LDS banks

__device__ __forceinline__ v16h load_frag(const half_t* __restrict__ p)
{
    v8h lo = *(const v8h*)(p);
    v8h hi = *(const v8h*)(p + 16);
    return __builtin_shufflevector(lo, hi,
           0,1,2,3,4,5,6,7,8,9,10,11,12,13,14,15);
}

// ---------------------------------------------------------------------------
// WMMA GEMM: out[M,N] = A[M,K](f16) * Bt[N,K](f16, pre-transposed) + bias
//            (+ residual) (RELU?) -> f32 and/or f16 outputs.
// Block = 256 threads = 8 waves; wave tile = 32(M) x 64(N); block = 256 x 64.
// N must be a multiple of 64 (caller pads Bt/bias); M multiple of 256.
// B panel staged through LDS in 64x256 chunks. Per K-step: 2 A frags (global),
// 4 B frags (LDS) preloaded, then 8 independent WMMAs issue back-to-back so
// each fragment is reused (A x4, B x2) and DS waits are partial, not drains.
// ---------------------------------------------------------------------------
template<int K, bool RELU>
__global__ void __launch_bounds__(256) gemm_f16_wmma(
    const half_t* __restrict__ A, const half_t* __restrict__ Bt,
    const float* __restrict__ bias, const float* __restrict__ residual,
    float* __restrict__ outF, half_t* __restrict__ outH, int N)
{
    __shared__ half_t smB[64 * SMB_STRIDE];

    const int wave = threadIdx.x >> 5;
    const int lane = threadIdx.x & 31;
    const int lrow = lane & 15;
    const int hsel = lane >> 4;          // K-interleave half select
    const int koff = hsel ? 8 : 0;

    const int m0 = blockIdx.y * 256 + wave * 32;
    const int n0 = blockIdx.x * 64;

    v8f acc[8];
#pragma unroll
    for (int t = 0; t < 8; ++t) acc[t] = (v8f){0.f,0.f,0.f,0.f,0.f,0.f,0.f,0.f};

    const half_t* arow0 = A + (size_t)(m0 + lrow) * K + koff;
    const half_t* arow1 = A + (size_t)(m0 + 16 + lrow) * K + koff;

    for (int kc = 0; kc < K; kc += 256) {
        // ---- stage B[n0..n0+63][kc..kc+255] into LDS (2048 x v8h chunks) ----
#pragma unroll
        for (int i = 0; i < 8; ++i) {
            int idx = i * 256 + threadIdx.x;
            int r = idx >> 5;
            int c = (idx & 31) * 8;
            *(v8h*)(&smB[r * SMB_STRIDE + c]) =
                *(const v8h*)(Bt + (size_t)(n0 + r) * K + kc + c);
        }
        __syncthreads();

        // ---- 8 K-steps of 32; preload 4 B frags, then 8 WMMAs --------------
#pragma unroll
        for (int kk = 0; kk < 256; kk += 32) {
            v16h a0 = load_frag(arow0 + kc + kk);
            v16h a1 = load_frag(arow1 + kc + kk);
            v16h b[4];
#pragma unroll
            for (int t = 0; t < 4; ++t)
                b[t] = load_frag(&smB[(t * 16 + lrow) * SMB_STRIDE + kk + koff]);
#pragma unroll
            for (int t = 0; t < 4; ++t)
                acc[t] = __builtin_amdgcn_wmma_f32_16x16x32_f16(
                    false, a0, false, b[t], (short)0, acc[t], false, false);
#pragma unroll
            for (int t = 0; t < 4; ++t)
                acc[4 + t] = __builtin_amdgcn_wmma_f32_16x16x32_f16(
                    false, a1, false, b[t], (short)0, acc[4 + t], false, false);
        }
        __syncthreads();
    }

#pragma unroll
    for (int half = 0; half < 2; ++half) {
#pragma unroll
        for (int t = 0; t < 4; ++t) {
            int col = n0 + t * 16 + lrow;
            float bv = bias ? bias[col] : 0.f;
#pragma unroll
            for (int r = 0; r < 8; ++r) {
                int row = m0 + half * 16 + hsel * 8 + r;
                float v = acc[half * 4 + t][r] + bv;
                if (residual) v += residual[(size_t)row * N + col];
                if (RELU) v = v > 0.f ? v : 0.f;
                size_t idx = (size_t)row * N + col;
                if (outF) outF[idx] = v;
                if (outH) outH[idx] = (half_t)v;
            }
        }
    }
}

// ---------------------------------------------------------------------------
// Deformable sampling + softmax. One wave per (q, head); lane = head-dim chan.
// v: [Lnum, QN, 256] f32; offaw row stride Ncat: cols [0,Noff) offsets
// (h,l,p,2), cols [Noff, Noff+NH*Lnum*4) attention logits (h, l*NP+p).
// ---------------------------------------------------------------------------
__device__ __forceinline__ float bicorner(const float* __restrict__ vl,
                                          int xi, int yi, int c)
{
    bool valid = (xi >= 0) & (xi < WGRID) & (yi >= 0) & (yi < HGRID);
    int xc = xi < 0 ? 0 : (xi > WGRID - 1 ? WGRID - 1 : xi);
    int yc = yi < 0 ? 0 : (yi > HGRID - 1 ? HGRID - 1 : yi);
    float val = vl[(size_t)(yc * WGRID + xc) * CN + c];
    return valid ? val : 0.f;
}

__global__ void __launch_bounds__(256) deform_sample(
    const float* __restrict__ v, const float* __restrict__ offaw,
    const float* __restrict__ ref2d, half_t* __restrict__ attnH,
    int Lnum, int Ncat, int Noff)
{
    const int wid  = blockIdx.x * 8 + (threadIdx.x >> 5);
    const int lane = threadIdx.x & 31;
    const int q = wid >> 3;
    const int h = wid & 7;

    const float* row = offaw + (size_t)q * Ncat;
    const int npts = Lnum * 4;

    float wts[8];
    float mx = -1e30f;
    for (int i = 0; i < npts; ++i) {
        wts[i] = row[Noff + h * npts + i];
        mx = fmaxf(mx, wts[i]);
    }
    float s = 0.f;
    for (int i = 0; i < npts; ++i) { wts[i] = expf(wts[i] - mx); s += wts[i]; }
    float inv = 1.f / s;

    const int c = h * 32 + lane;
    float acc = 0.f;
    for (int l = 0; l < Lnum; ++l) {
        float rx = ref2d[((size_t)q * 2 + l) * 2 + 0];
        float ry = ref2d[((size_t)q * 2 + l) * 2 + 1];
        const float* vl = v + (size_t)l * QN * CN;
        for (int p = 0; p < 4; ++p) {
            float ox = row[((h * Lnum + l) * 4 + p) * 2 + 0];
            float oy = row[((h * Lnum + l) * 4 + p) * 2 + 1];
            float x = (rx + ox * (1.f / (float)WGRID)) * (float)WGRID - 0.5f;
            float y = (ry + oy * (1.f / (float)HGRID)) * (float)HGRID - 0.5f;
            float x0f = floorf(x), y0f = floorf(y);
            int x0 = (int)x0f, y0 = (int)y0f;
            float tx = x - x0f, ty = y - y0f;
            float c00 = bicorner(vl, x0,     y0,     c);
            float c10 = bicorner(vl, x0 + 1, y0,     c);
            float c01 = bicorner(vl, x0,     y0 + 1, c);
            float c11 = bicorner(vl, x0 + 1, y0 + 1, c);
            float top = c00 * (1.f - tx) + c10 * tx;
            float bot = c01 * (1.f - tx) + c11 * tx;
            acc += wts[l * 4 + p] * inv * (top * (1.f - ty) + bot * ty);
        }
    }
    attnH[(size_t)q * CN + c] = (half_t)acc;
}

// ---------------------------------------------------------------------------
// LayerNorm over C=256: one wave per row (8 elems/lane), shfl_xor reductions.
// ---------------------------------------------------------------------------
__global__ void __launch_bounds__(256) layernorm_rows(
    const float* __restrict__ x, const float* __restrict__ g,
    const float* __restrict__ b, float* __restrict__ outF,
    half_t* __restrict__ outH)
{
    const int row  = blockIdx.x * 8 + (threadIdx.x >> 5);
    const int lane = threadIdx.x & 31;
    const float* xr = x + (size_t)row * CN;

    float vals[8];
    float s = 0.f;
#pragma unroll
    for (int i = 0; i < 8; ++i) { vals[i] = xr[i * 32 + lane]; s += vals[i]; }
#pragma unroll
    for (int o = 16; o; o >>= 1) s += __shfl_xor(s, o, 32);
    float mean = s * (1.f / (float)CN);

    float vv = 0.f;
#pragma unroll
    for (int i = 0; i < 8; ++i) { float d = vals[i] - mean; vv += d * d; }
#pragma unroll
    for (int o = 16; o; o >>= 1) vv += __shfl_xor(vv, o, 32);
    float rs = rsqrtf(vv * (1.f / (float)CN) + 1e-5f);

#pragma unroll
    for (int i = 0; i < 8; ++i) {
        int cc = i * 32 + lane;
        float o = (vals[i] - mean) * rs * g[cc] + b[cc];
        outF[(size_t)row * CN + cc] = o;
        if (outH) outH[(size_t)row * CN + cc] = (half_t)o;
    }
}

// out[i] = f16(a[i] + b[i])  (b nullable)
__global__ void __launch_bounds__(256) addcvt_f16(
    const float* __restrict__ a, const float* __restrict__ b,
    half_t* __restrict__ out, int n)
{
    int i = blockIdx.x * 256 + threadIdx.x;
    if (i < n) out[i] = (half_t)(b ? (a[i] + b[i]) : a[i]);
}

// Bt[(rowOff+n)*K + k] = (f16) W[k*N + n]
__global__ void __launch_bounds__(256) transpose_cvt(
    const float* __restrict__ W, half_t* __restrict__ Bt,
    int K, int N, int rowOff)
{
    int i = blockIdx.x * 256 + threadIdx.x;
    if (i >= K * N) return;
    int n = i / K, k = i % K;
    Bt[(size_t)(rowOff + n) * K + k] = (half_t)W[(size_t)k * N + n];
}

// ---------------------------------------------------------------------------
extern "C" void kernel_launch(void* const* d_in, const int* in_sizes, int n_in,
                              void* d_out, int out_size, void* d_ws, size_t ws_size,
                              hipStream_t stream)
{
    (void)in_sizes; (void)n_in; (void)out_size; (void)ws_size;
    const float* query   = (const float*)d_in[0];
    const float* qpos    = (const float*)d_in[1];
    const float* value   = (const float*)d_in[2];
    const float* ref2d   = (const float*)d_in[3];
    const void*  sshapes = d_in[4];
    const float* sa_off_w = (const float*)d_in[5];
    const float* sa_off_b = (const float*)d_in[6];
    const float* sa_aw_w  = (const float*)d_in[7];
    const float* sa_aw_b  = (const float*)d_in[8];
    const float* sa_vp_w  = (const float*)d_in[9];
    const float* sa_vp_b  = (const float*)d_in[10];
    const float* sa_op_w  = (const float*)d_in[11];
    const float* sa_op_b  = (const float*)d_in[12];
    const float* ca_off_w = (const float*)d_in[13];
    const float* ca_off_b = (const float*)d_in[14];
    const float* ca_aw_w  = (const float*)d_in[15];
    const float* ca_aw_b  = (const float*)d_in[16];
    const float* ca_vp_w  = (const float*)d_in[17];
    const float* ca_vp_b  = (const float*)d_in[18];
    const float* ca_op_w  = (const float*)d_in[19];
    const float* ca_op_b  = (const float*)d_in[20];
    const float* ffn_w1   = (const float*)d_in[21];
    const float* ffn_b1   = (const float*)d_in[22];
    const float* ffn_w2   = (const float*)d_in[23];
    const float* ffn_b2   = (const float*)d_in[24];
    const float* ln1_g = (const float*)d_in[25];
    const float* ln1_b = (const float*)d_in[26];
    const float* ln2_g = (const float*)d_in[27];
    const float* ln2_b = (const float*)d_in[28];
    const float* ln3_g = (const float*)d_in[29];
    const float* ln3_b = (const float*)d_in[30];

    const size_t QC = (size_t)QN * CN;

    // --- workspace layout -------------------------------------------------
    size_t off = 0;
    auto walloc = [&](size_t bytes) -> size_t {
        size_t o = off; off = (off + bytes + 255) & ~(size_t)255; return o;
    };
    char* ws = (char*)d_ws;
    half_t* h_qA   = (half_t*)(ws + walloc(QC * 2));          // query (f16)
    half_t* h_qin  = (half_t*)(ws + walloc(QC * 2));          // q(+pos) f16, reused
    half_t* h_vA   = (half_t*)(ws + walloc(2 * QC * 2));      // value f16
    half_t* h_attn = (half_t*)(ws + walloc(QC * 2));          // sampled attn f16
    half_t* h_ffn  = (half_t*)(ws + walloc((size_t)QN * 1024 * 2)); // ffn hidden f16
    half_t* w_savp = (half_t*)(ws + walloc(256 * 256 * 2));
    half_t* w_saoa = (half_t*)(ws + walloc(128 * 256 * 2));   // off(64)|aw(32)|pad(32)
    half_t* w_saop = (half_t*)(ws + walloc(256 * 256 * 2));
    half_t* w_cavp = (half_t*)(ws + walloc(256 * 256 * 2));
    half_t* w_caoa = (half_t*)(ws + walloc(192 * 256 * 2));   // off(128)|aw(64)
    half_t* w_caop = (half_t*)(ws + walloc(256 * 256 * 2));
    half_t* w_ffn1 = (half_t*)(ws + walloc(1024 * 256 * 2));
    half_t* w_ffn2 = (half_t*)(ws + walloc(256 * 1024 * 2));
    float*  b_sa   = (float*)(ws + walloc(128 * 4));
    float*  b_ca   = (float*)(ws + walloc(192 * 4));
    float*  f_vsa  = (float*)(ws + walloc(QC * 4));
    float*  f_vca  = (float*)(ws + walloc(2 * QC * 4));
    float*  f_oa   = (float*)(ws + walloc((size_t)QN * 192 * 4));
    float*  f_pre  = (float*)(ws + walloc(QC * 4));           // pre-LN
    float*  f_q    = (float*)(ws + walloc(QC * 4));           // post-LN f32

    // --- prep: weight transpose+convert, bias concat, zero pads ------------
    auto tgrid = [](int K, int N) { return dim3((unsigned)((K * N + 255) / 256)); };
    transpose_cvt<<<tgrid(256, 256),  256, 0, stream>>>(sa_vp_w, w_savp, 256, 256, 0);
    transpose_cvt<<<tgrid(256, 64),   256, 0, stream>>>(sa_off_w, w_saoa, 256, 64, 0);
    transpose_cvt<<<tgrid(256, 32),   256, 0, stream>>>(sa_aw_w,  w_saoa, 256, 32, 64);
    transpose_cvt<<<tgrid(256, 256),  256, 0, stream>>>(sa_op_w, w_saop, 256, 256, 0);
    transpose_cvt<<<tgrid(256, 256),  256, 0, stream>>>(ca_vp_w, w_cavp, 256, 256, 0);
    transpose_cvt<<<tgrid(256, 128),  256, 0, stream>>>(ca_off_w, w_caoa, 256, 128, 0);
    transpose_cvt<<<tgrid(256, 64),   256, 0, stream>>>(ca_aw_w,  w_caoa, 256, 64, 128);
    transpose_cvt<<<tgrid(256, 256),  256, 0, stream>>>(ca_op_w, w_caop, 256, 256, 0);
    transpose_cvt<<<tgrid(256, 1024), 256, 0, stream>>>(ffn_w1, w_ffn1, 256, 1024, 0);
    transpose_cvt<<<tgrid(1024, 256), 256, 0, stream>>>(ffn_w2, w_ffn2, 1024, 256, 0);
    hipMemsetAsync(w_saoa + 96 * 256, 0, 32 * 256 * 2, stream);   // pad rows 96..127
    hipMemcpyAsync(b_sa,       sa_off_b,  64 * 4, hipMemcpyDeviceToDevice, stream);
    hipMemcpyAsync(b_sa + 64,  sa_aw_b,   32 * 4, hipMemcpyDeviceToDevice, stream);
    hipMemsetAsync(b_sa + 96,  0,         32 * 4, stream);        // pad bias
    hipMemcpyAsync(b_ca,       ca_off_b, 128 * 4, hipMemcpyDeviceToDevice, stream);
    hipMemcpyAsync(b_ca + 128, ca_aw_b,   64 * 4, hipMemcpyDeviceToDevice, stream);

    int nQC = (int)QC;
    addcvt_f16<<<(nQC + 255) / 256, 256, 0, stream>>>(query, nullptr, h_qA, nQC);
    addcvt_f16<<<(nQC + 255) / 256, 256, 0, stream>>>(query, qpos, h_qin, nQC);
    addcvt_f16<<<(2 * nQC + 255) / 256, 256, 0, stream>>>(value, nullptr, h_vA, 2 * nQC);

    auto ggrid = [](int M, int N) { return dim3((unsigned)(N / 64), (unsigned)(M / 256)); };

    // --- self-attention (deformable, L=1) ---------------------------------
    gemm_f16_wmma<256, false><<<ggrid(QN, 256), 256, 0, stream>>>(
        h_qA, w_savp, sa_vp_b, nullptr, f_vsa, nullptr, 256);
    gemm_f16_wmma<256, false><<<ggrid(QN, 128), 256, 0, stream>>>(
        h_qin, w_saoa, b_sa, nullptr, f_oa, nullptr, 128);
    deform_sample<<<QN, 256, 0, stream>>>(f_vsa, f_oa, ref2d, h_attn, 1, 128, 64);
    gemm_f16_wmma<256, false><<<ggrid(QN, 256), 256, 0, stream>>>(
        h_attn, w_saop, sa_op_b, query, f_pre, nullptr, 256);
    layernorm_rows<<<QN / 8, 256, 0, stream>>>(f_pre, ln1_g, ln1_b, f_q, nullptr);
    addcvt_f16<<<(nQC + 255) / 256, 256, 0, stream>>>(f_q, qpos, h_qin, nQC);

    // --- cross-attention (deformable, L=2) --------------------------------
    gemm_f16_wmma<256, false><<<ggrid(2 * QN, 256), 256, 0, stream>>>(
        h_vA, w_cavp, ca_vp_b, nullptr, f_vca, nullptr, 256);
    gemm_f16_wmma<256, false><<<ggrid(QN, 192), 256, 0, stream>>>(
        h_qin, w_caoa, b_ca, nullptr, f_oa, nullptr, 192);
    deform_sample<<<QN, 256, 0, stream>>>(f_vca, f_oa, ref2d, h_attn, 2, 192, 128);
    gemm_f16_wmma<256, false><<<ggrid(QN, 256), 256, 0, stream>>>(
        h_attn, w_caop, ca_op_b, f_q, f_pre, nullptr, 256);
    layernorm_rows<<<QN / 8, 256, 0, stream>>>(f_pre, ln2_g, ln2_b, f_q, h_qin);

    // --- FFN ---------------------------------------------------------------
    gemm_f16_wmma<256, true><<<ggrid(QN, 1024), 256, 0, stream>>>(
        h_qin, w_ffn1, ffn_b1, nullptr, nullptr, h_ffn, 1024);
    gemm_f16_wmma<1024, false><<<ggrid(QN, 256), 256, 0, stream>>>(
        h_ffn, w_ffn2, ffn_b2, f_q, f_pre, nullptr, 256);
    layernorm_rows<<<QN / 8, 256, 0, stream>>>(f_pre, ln3_g, ln3_b, (float*)d_out, nullptr);

    // --- tuple tail: (query_pos, value, ref_2d, spatial_shapes) ------------
    float* outf = (float*)d_out;
    hipMemcpyAsync(outf + QC,     qpos,  QC * 4,             hipMemcpyDeviceToDevice, stream);
    hipMemcpyAsync(outf + 2 * QC, value, 2 * QC * 4,         hipMemcpyDeviceToDevice, stream);
    hipMemcpyAsync(outf + 4 * QC, ref2d, (size_t)QN * 4 * 4, hipMemcpyDeviceToDevice, stream);
    hipMemcpyAsync(outf + 4 * QC + (size_t)QN * 4, sshapes, 4 * 4,
                   hipMemcpyDeviceToDevice, stream);
}
